// Head_69535520522793
// MI455X (gfx1250) — matrix operand order repaired and checked by reading, not compiled, not run on hardware
//
#include <hip/hip_runtime.h>
#include <math.h>

typedef __attribute__((ext_vector_type(16))) _Float16 v16h;
typedef __attribute__((ext_vector_type(8)))  _Float16 v8h;
typedef __attribute__((ext_vector_type(8)))  float    v8f;

#define B_  512
#define T_  256
#define C_  384
#define H_  64
#define BT_ (B_*T_)

static __device__ __forceinline__ v8f wmma_f16(v16h a, v16h b, v8f c) {
  // (neg_a, A, neg_b, B, c_mod, C, reuse_a, reuse_b)
  return __builtin_amdgcn_wmma_f32_16x16x32_f16(false, a, false, b, (short)0, c, false, false);
}

// ---------------------------------------------------------------------------
// Kernel 1: QKV projection.  [BT, 384] x [384, 64] (x3) with f16 WMMA, f32 acc.
// Block = 256 threads = 8 waves, each wave computes 16 rows x 64 cols for q,k,v.
// Weights live transposed in LDS as [n][k] (f16) so B-fragments are contiguous.
// B-fragments are staged in register groups of 6 so one s_wait_dscnt covers
// six WMMAs and no WAR hazard NOPs are needed between WMMA and the next loads.
// ---------------------------------------------------------------------------
__global__ __launch_bounds__(256) void qkv_proj_kernel(
    const float* __restrict__ x,
    const float* __restrict__ Wk, const float* __restrict__ Wq, const float* __restrict__ Wv,
    _Float16* __restrict__ qh, _Float16* __restrict__ kh, _Float16* __restrict__ vh)
{
  __shared__ _Float16 wlds[3][H_][C_];   // 3 * 64 * 384 * 2B = 144 KB

  const int tid = threadIdx.x;
  const float* wsrc[3] = {Wq, Wk, Wv};
  for (int w = 0; w < 3; ++w) {
    const float* src = wsrc[w];
    for (int i = tid; i < C_ * H_; i += 256) {
      const int k = i >> 6;        // i = k*64 + n  (coalesced global read)
      const int n = i & 63;
      wlds[w][n][k] = (_Float16)src[i];
    }
  }
  __syncthreads();

  const int wave = tid >> 5, lane = tid & 31;
  const int ln = lane & 15, hi = lane >> 4;
  const int row0 = blockIdx.x * 128 + wave * 16;
  const float* xrow = x + (size_t)(row0 + ln) * C_;

  // Per-lane base pointers into the 12 (weight, ntile) B columns; k varies per step.
  const _Float16* bcol[12];
#pragma unroll
  for (int w = 0; w < 3; ++w)
#pragma unroll
    for (int t = 0; t < 4; ++t)
      bcol[w * 4 + t] = &wlds[w][t * 16 + ln][hi * 16];

  v8f acc[12] = {};   // [weight*4 + ntile], 16x16 f32 each

#pragma unroll 1
  for (int ks = 0; ks < C_; ks += 32) {
    // A fragment (16x32 f16): lanes 0-15 hold K=ks+{0..15}, lanes 16-31 K=ks+{16..31}
    const float4 g0 = *(const float4*)(xrow + ks + hi * 16);
    const float4 g1 = *(const float4*)(xrow + ks + hi * 16 + 4);
    const float4 g2 = *(const float4*)(xrow + ks + hi * 16 + 8);
    const float4 g3 = *(const float4*)(xrow + ks + hi * 16 + 12);
    v16h a;
    a[0]=(_Float16)g0.x;  a[1]=(_Float16)g0.y;  a[2]=(_Float16)g0.z;  a[3]=(_Float16)g0.w;
    a[4]=(_Float16)g1.x;  a[5]=(_Float16)g1.y;  a[6]=(_Float16)g1.z;  a[7]=(_Float16)g1.w;
    a[8]=(_Float16)g2.x;  a[9]=(_Float16)g2.y;  a[10]=(_Float16)g2.z; a[11]=(_Float16)g2.w;
    a[12]=(_Float16)g3.x; a[13]=(_Float16)g3.y; a[14]=(_Float16)g3.z; a[15]=(_Float16)g3.w;

    // Two staged groups of 6: issue all LDS loads, then all WMMAs.
#pragma unroll
    for (int g = 0; g < 2; ++g) {
      v16h bf[6];
#pragma unroll
      for (int i = 0; i < 6; ++i)
        bf[i] = *(const v16h*)(bcol[g * 6 + i] + ks);
#pragma unroll
      for (int i = 0; i < 6; ++i)
        acc[g * 6 + i] = wmma_f16(a, bf[i], acc[g * 6 + i]);
    }
  }

  _Float16* outs[3] = {qh, kh, vh};
#pragma unroll
  for (int w = 0; w < 3; ++w) {
#pragma unroll
    for (int t = 0; t < 4; ++t) {
#pragma unroll
      for (int v = 0; v < 8; ++v) {
        const int r = row0 + v + 8 * hi;           // D layout: M = vgpr + 8*(lane>=16)
        outs[w][(size_t)r * H_ + t * 16 + ln] = (_Float16)acc[w * 4 + t][v];
      }
    }
  }
}

// ---------------------------------------------------------------------------
// Kernel 2: causal flash-attention.  One block per (batch, 128-query chunk).
// K row-major + V transposed in LDS; online softmax; all matmuls via WMMA f16.
// Fragment loads are grouped ahead of the WMMA bursts to minimize dscnt waits.
// ---------------------------------------------------------------------------
__global__ __launch_bounds__(256) void attn_kernel(
    const _Float16* __restrict__ qh, const _Float16* __restrict__ kh,
    const _Float16* __restrict__ vh, float* __restrict__ out)
{
  __shared__ _Float16 klds[T_][H_];      // [key][h]   32 KB
  __shared__ _Float16 vt[H_][T_];        // [h][key]   32 KB (transposed)
  __shared__ _Float16 plds[8][16][32];   // per-wave P staging, 8 KB

  const int tid = threadIdx.x;
  const int b  = blockIdx.x >> 1;
  const int qc = (blockIdx.x & 1) * 128;
  const size_t base = (size_t)b * T_ * H_;

  // Load K (vectorized) and V (transposed) for this batch into LDS.
  for (int i = tid; i < T_ * H_ / 8; i += 256)
    *(v8h*)((_Float16*)klds + i * 8) = *(const v8h*)(kh + base + i * 8);
  for (int i = tid; i < T_ * H_; i += 256) {
    const int key = i >> 6, h = i & 63;
    vt[h][key] = vh[base + i];
  }
  __syncthreads();

  const int wave = tid >> 5, lane = tid & 31;
  const int ln = lane & 15, hi = lane >> 4;
  const int qrow0 = qc + wave * 16;

  // Q as two 16x32 A-fragments (h = 0..31, 32..63)
  const _Float16* qrow = qh + base + (size_t)(qrow0 + ln) * H_;
  v16h aq[2];
#pragma unroll
  for (int s = 0; s < 2; ++s) {
    const v8h lo  = *(const v8h*)(qrow + s * 32 + hi * 16);
    const v8h hi8 = *(const v8h*)(qrow + s * 32 + hi * 16 + 8);
    aq[s] = __builtin_shufflevector(lo, hi8, 0,1,2,3,4,5,6,7,8,9,10,11,12,13,14,15);
  }

  v8f o[4] = {};
  float mrow[8], lrow[8];
#pragma unroll
  for (int v = 0; v < 8; ++v) { mrow[v] = -1e30f; lrow[v] = 0.0f; }

  const float scale = 0.05103103630798288f;  // 384^-0.5 (reference scales by C, not H)

  const int kend = qrow0 + 16;               // causal: keys <= last query of this wave
#pragma unroll 1
  for (int j0 = 0; j0 < kend; j0 += 32) {
    // ---- S = Q K^T for 32 keys (two 16-key subtiles); stage all 4 K-fragments ----
    v16h bk[4];
#pragma unroll
    for (int s = 0; s < 2; ++s) {
      bk[s * 2 + 0] = *(const v16h*)&klds[j0 + s * 16 + ln][hi * 16];        // h 0..31
      bk[s * 2 + 1] = *(const v16h*)&klds[j0 + s * 16 + ln][32 + hi * 16];   // h 32..63
    }
    v8f sacc[2] = {};
#pragma unroll
    for (int s = 0; s < 2; ++s) {
      sacc[s] = wmma_f16(aq[0], bk[s * 2 + 0], sacc[s]);
      sacc[s] = wmma_f16(aq[1], bk[s * 2 + 1], sacc[s]);
    }
    float sv[2][8];
#pragma unroll
    for (int s = 0; s < 2; ++s) {
#pragma unroll
      for (int v = 0; v < 8; ++v) {
        const int qg = qrow0 + v + 8 * hi;
        const int kg = j0 + s * 16 + ln;
        sv[s][v] = (kg <= qg) ? (float)sacc[s][v] * scale : -1e30f;
      }
    }

    // ---- online softmax (rows spread: row = v + 8*hi, cols across 16 lanes) ----
#pragma unroll
    for (int v = 0; v < 8; ++v) {
      float mx = fmaxf(sv[0][v], sv[1][v]);
      for (int off = 1; off < 16; off <<= 1) mx = fmaxf(mx, __shfl_xor(mx, off, 16));
      const float mnew  = fmaxf(mrow[v], mx);
      const float alpha = __expf(mrow[v] - mnew);
      mrow[v] = mnew;
      const float p0 = __expf(sv[0][v] - mnew);
      const float p1 = __expf(sv[1][v] - mnew);
      float ps = p0 + p1;
      for (int off = 1; off < 16; off <<= 1) ps += __shfl_xor(ps, off, 16);
      lrow[v] = lrow[v] * alpha + ps;
#pragma unroll
      for (int t = 0; t < 4; ++t) o[t][v] *= alpha;
      plds[wave][v + 8 * hi][ln]      = (_Float16)p0;   // D-layout -> [query][key]
      plds[wave][v + 8 * hi][16 + ln] = (_Float16)p1;
    }

    // ---- re-lay out P as a 16x32 A-fragment (same-wave LDS round trip) ----
    const v8h plo = *(const v8h*)&plds[wave][ln][hi * 16];
    const v8h phi = *(const v8h*)&plds[wave][ln][hi * 16 + 8];
    const v16h pa = __builtin_shufflevector(plo, phi, 0,1,2,3,4,5,6,7,8,9,10,11,12,13,14,15);

    // ---- O += P V : stage all 4 V-fragments, then 4 WMMAs ----
    v16h bv[4];
#pragma unroll
    for (int t = 0; t < 4; ++t)
      bv[t] = *(const v16h*)&vt[t * 16 + ln][j0 + hi * 16];
#pragma unroll
    for (int t = 0; t < 4; ++t)
      o[t] = wmma_f16(pa, bv[t], o[t]);
  }

  // ---- finalize: divide by row sums, store fp32 ----
#pragma unroll
  for (int v = 0; v < 8; ++v) {
    const float inv = 1.0f / lrow[v];
    const int qg = qrow0 + v + 8 * hi;
    float* orow = out + base + (size_t)qg * H_;
#pragma unroll
    for (int t = 0; t < 4; ++t) orow[t * 16 + ln] = o[t][v] * inv;
  }
}

extern "C" void kernel_launch(void* const* d_in, const int* in_sizes, int n_in,
                              void* d_out, int out_size, void* d_ws, size_t ws_size,
                              hipStream_t stream) {
  const float* x  = (const float*)d_in[0];
  const float* Wk = (const float*)d_in[1];
  const float* Wq = (const float*)d_in[2];
  const float* Wv = (const float*)d_in[3];

  _Float16* qh = (_Float16*)d_ws;                 // 3 * BT * 64 f16 = ~50 MB scratch
  _Float16* kh = qh + (size_t)BT_ * H_;
  _Float16* vh = kh + (size_t)BT_ * H_;

  qkv_proj_kernel<<<BT_ / 128, 256, 0, stream>>>(x, Wk, Wq, Wv, qh, kh, vh);
  attn_kernel<<<B_ * 2, 256, 0, stream>>>(qh, kh, vh, (float*)d_out);
}
